// Refine_Network_31104153157904
// MI455X (gfx1250) — matrix operand order repaired and checked by reading, not compiled, not run on hardware
//
#include <hip/hip_runtime.h>

// ---------------- constants baked from the reference ----------------
namespace cfg {
constexpr int L      = 1024;
constexpr int KNB    = 64;     // top_k
constexpr int DPAIR  = 128;
constexpr int CE     = 32;
constexpr int RTOT   = 674;    // 16*32 + 16*3 + 3*32 + 3*3 + 3*3
constexpr int RPAD   = 688;    // 43 * 16
constexpr int F2PAD  = 96;     // 69 (32+36+1) padded to 3*32
}

typedef __bf16 bf16;
typedef __attribute__((ext_vector_type(16))) __bf16 v16bf;
typedef __attribute__((ext_vector_type(8)))  float  v8f;

// Intra-wave LDS ordering: DS ops are in-order per wave in hardware; this
// fence only has to stop the compiler from reordering LDS accesses. It emits
// s_wait_dscnt but no s_barrier, so waves in a block do not lock-step.
#define WAVE_LDS_SYNC() __builtin_amdgcn_fence(__ATOMIC_ACQ_REL, "workgroup")

__device__ __forceinline__ float wsum32(float x) {
  x += __shfl_xor(x, 1, 32);
  x += __shfl_xor(x, 2, 32);
  x += __shfl_xor(x, 4, 32);
  x += __shfl_xor(x, 8, 32);
  x += __shfl_xor(x, 16, 32);
  return x;
}

__device__ __forceinline__ v8f vzero8() {
  v8f z = {0.f, 0.f, 0.f, 0.f, 0.f, 0.f, 0.f, 0.f};
  return z;
}

__device__ __forceinline__ v8f wmma_bf16(v16bf a, v16bf b, v8f c) {
  return __builtin_amdgcn_wmma_f32_16x16x32_bf16(false, a, false, b, (short)0, c,
                                                 false, false);
}

// ---- fragment-major storage -------------------------------------------------
// A wave32 16-bit fragment: lane l = m + 16*half owns 16 contiguous bf16.
// A-matrix element (row r, K index c) lives at:
//   kb = c>>5, kk = c&31, half = kk>>4, e = kk&15
//   flat = ((kb*32) + r + 16*half)*16 + e
// B-matrix (KxN weights): lane l holds K = 16*(l>>4)+e of column n=(l&15); we
// stage weights pre-swizzled so the same flat layout applies per n-tile slot.
__device__ __forceinline__ int frag_idx(int r, int c) {
  const int kb = c >> 5, kk = c & 31;
  const int half = kk >> 4;
  const int e = kk & 15;
  return ((kb * 32) + r + 16 * half) * 16 + e;
}

// One v16bf load (lowers to 2x ds_load_b128) per fragment.
__device__ __forceinline__ v16bf load_frag(const bf16* base, int slot, int lane) {
  return *(const v16bf*)(base + (slot * 32 + lane) * 16);
}

// C/D layout: VGPR v, lanes 0-15 -> row v, lanes 16-31 -> row v+8; col = lane&15
__device__ __forceinline__ void dump_acc(float* S, int lane, int colOff, int stride,
                                         v8f acc) {
  const int half = lane >> 4, c = (lane & 15) + colOff;
#pragma unroll
  for (int v = 0; v < 8; ++v) S[(v + 8 * half) * stride + c] = acc[v];
}

__device__ __forceinline__ void dump_acc_bias(float* S, int lane, int colOff,
                                              int stride, v8f acc, const float* bias) {
  const int half = lane >> 4, c = (lane & 15) + colOff;
#pragma unroll
  for (int v = 0; v < 8; ++v) S[(v + 8 * half) * stride + c] = acc[v] + bias[c];
}

// Row-wise LayerNorm over 32 cols of a 16x32 f32 LDS tile -> bf16 fragment tile
__device__ __forceinline__ void ln32_rows_to_frag(const float* S, const float* g,
                                                  const float* b, bf16* out, int lane) {
#pragma unroll
  for (int r = 0; r < 16; ++r) {
    float x = S[r * 32 + lane];
    float mu = wsum32(x) * (1.f / 32.f);
    float xm = x - mu;
    float var = wsum32(xm * xm) * (1.f / 32.f);
    float rs = rsqrtf(var + 1e-5f);
    out[frag_idx(r, lane)] = (bf16)(xm * rs * g[lane] + b[lane]);
  }
}

// ======================= K1: node features (L x 32) =======================
__global__ __launch_bounds__(32) void node_kernel(
    const float* __restrict__ msa, const float* __restrict__ state,
    const float* __restrict__ seq1hot, const float* __restrict__ g_msa,
    const float* __restrict__ b_msa, const float* __restrict__ g_state,
    const float* __restrict__ b_state, const float* __restrict__ Wx,
    const float* __restrict__ bx, const float* __restrict__ g_node,
    const float* __restrict__ b_node, float* __restrict__ node) {
  const int i = blockIdx.x, lane = threadIdx.x;
  __shared__ float xf[101];
  float a = msa[i * 64 + lane], b = msa[i * 64 + 32 + lane];
  float mu = wsum32(a + b) * (1.f / 64.f);
  float ya = a - mu, yb = b - mu;
  float var = wsum32(ya * ya + yb * yb) * (1.f / 64.f);
  float rs = rsqrtf(var + 1e-5f);
  xf[lane] = ya * rs * g_msa[lane] + b_msa[lane];
  xf[32 + lane] = yb * rs * g_msa[32 + lane] + b_msa[32 + lane];
  if (lane < 21) xf[64 + lane] = seq1hot[i * 21 + lane];
  float st = (lane < 16) ? state[i * 16 + lane] : 0.f;
  float ms = wsum32(st) * (1.f / 16.f);
  float ds = (lane < 16) ? (st - ms) : 0.f;
  float vs = wsum32(ds * ds) * (1.f / 16.f);
  float rss = rsqrtf(vs + 1e-5f);
  if (lane < 16) xf[85 + lane] = ds * rss * g_state[lane] + b_state[lane];
  __syncthreads();
  float y = bx[lane];
  for (int c = 0; c < 101; ++c) y += xf[c] * Wx[c * 32 + lane];
  float my = wsum32(y) * (1.f / 32.f);
  float ym = y - my;
  float vy = wsum32(ym * ym) * (1.f / 32.f);
  float ry = rsqrtf(vy + 1e-5f);
  node[i * 32 + lane] = ym * ry * g_node[lane] + b_node[lane];
}

// ======================= K2: exact top-k (jax tie-break) ==================
__global__ __launch_bounds__(256) void topk_kernel(const float* __restrict__ xyz,
                                                   int* __restrict__ Eidx) {
  constexpr int L = cfg::L;
  const int i = blockIdx.x, t = threadIdx.x;
  __shared__ float Drow[L];
  __shared__ float rv[256];
  __shared__ int ri[256];
  const float cx = xyz[(i * 3 + 1) * 3 + 0];
  const float cy = xyz[(i * 3 + 1) * 3 + 1];
  const float cz = xyz[(i * 3 + 1) * 3 + 2];
  for (int j = t; j < L; j += 256) {
    float dx = xyz[(j * 3 + 1) * 3 + 0] - cx;
    float dy = xyz[(j * 3 + 1) * 3 + 1] - cy;
    float dz = xyz[(j * 3 + 1) * 3 + 2] - cz;
    float D = sqrtf(fmaxf(dx * dx + dy * dy + dz * dz, 1e-8f));
    if (j == i) D += 999.9f;
    Drow[j] = D;
  }
  __syncthreads();
  for (int it = 0; it < cfg::KNB; ++it) {
    float best = 3.0e38f;
    int bj = L;
    for (int j = t; j < L; j += 256) {
      float v = Drow[j];
      if (v < best) { best = v; bj = j; }
    }
    rv[t] = best; ri[t] = bj;
    __syncthreads();
    for (int s = 128; s > 0; s >>= 1) {
      if (t < s) {
        float v = rv[t + s]; int vj = ri[t + s];
        if (v < rv[t] || (v == rv[t] && vj < ri[t])) { rv[t] = v; ri[t] = vj; }
      }
      __syncthreads();
    }
    if (t == 0) { Eidx[i * cfg::KNB + it] = ri[0]; Drow[ri[0]] = 3.4e38f; }
    __syncthreads();
  }
}

// ========== K3: gathered pair trunk + message contraction (WMMA) ==========
// block = 128 threads = 4 waves; block b handles residue i=b; wave w handles
// neighbors k = 16w .. 16w+15 (one 16-row WMMA tile).
__global__ __launch_bounds__(128) void pair_gemm_kernel(
    const float* __restrict__ pair, const float* __restrict__ xyz,
    const float* __restrict__ node, const int* __restrict__ Eidx,
    const float* __restrict__ g_pair, const float* __restrict__ b_pair,
    const float* __restrict__ We1, const float* __restrict__ be1,
    const float* __restrict__ g_e1, const float* __restrict__ b_e1,
    const float* __restrict__ We2, const float* __restrict__ be2,
    const float* __restrict__ g_e2, const float* __restrict__ b_e2,
    const float* __restrict__ Wr1, const float* __restrict__ br1,
    const float* __restrict__ Wr2, const float* __restrict__ br2,
    float* __restrict__ pm0, float* __restrict__ pm1) {
  using namespace cfg;
  // weights, fragment-major: [slot][lane][16] bf16, one v16bf load per fragment
  __shared__ __align__(32) bf16 sWe1f[4 * 2 * 32 * 16];   // [kb][nt]
  __shared__ __align__(32) bf16 sWe2f[3 * 2 * 32 * 16];   // [kb][nt]
  __shared__ __align__(32) bf16 sWr1f[2 * 32 * 16];       // [nt]
  __shared__ __align__(32) bf16 sWr2f[43 * 32 * 16];      // [ntile]
  __shared__ float sbe1[CE], sbe2[CE], sbr1[CE], sbr2[RPAD];
  __shared__ float sgp[DPAIR], sbp[DPAIR];
  __shared__ float sge1[CE], sbel1[CE], sge2[CE], sbel2[CE];
  // per-wave activation tiles, fragment-major
  __shared__ __align__(32) bf16 sA[4][4 * 32 * 16];       // 16x128
  __shared__ __align__(32) bf16 sF[4][3 * 32 * 16];       // 16x96
  __shared__ __align__(32) bf16 sWt[4][32 * 16];          // 16x32
  __shared__ __align__(32) bf16 sH[4][32 * 16];           // 16x32
  __shared__ float sS[4][16 * F2PAD];
  __shared__ float sf0[4][16 * CE];
  __shared__ float sf1[4][16 * 9];
  __shared__ float sdh[4][16 * 3];
  __shared__ float spj[4][16 * 3];
  __shared__ float sDij[4][16];
  __shared__ float snbr[4][16];
  __shared__ int sjb[4][16];

  const int t = threadIdx.x;
  const int i = blockIdx.x;
  const int wq = t >> 5, lane = t & 31;
  const int half = lane >> 4, r16 = lane & 15;

  // ---- stage weights into LDS, pre-swizzled to fragment layout ----
  for (int idx = t; idx < 4 * 2 * 32 * 16; idx += 128) {
    int e = idx & 15, ln = (idx >> 4) & 31, nt = (idx >> 9) & 1, kb = idx >> 10;
    int k = kb * 32 + 16 * (ln >> 4) + e, n = nt * 16 + (ln & 15);
    sWe1f[idx] = (bf16)We1[k * CE + n];
  }
  for (int idx = t; idx < 3 * 2 * 32 * 16; idx += 128) {
    int e = idx & 15, ln = (idx >> 4) & 31, nt = (idx >> 9) & 1, kb = idx >> 10;
    int k = kb * 32 + 16 * (ln >> 4) + e, n = nt * 16 + (ln & 15);
    sWe2f[idx] = (bf16)((k < 69) ? We2[k * CE + n] : 0.f);
  }
  for (int idx = t; idx < 2 * 32 * 16; idx += 128) {
    int e = idx & 15, ln = (idx >> 4) & 31, nt = idx >> 9;
    int k = 16 * (ln >> 4) + e, n = nt * 16 + (ln & 15);
    sWr1f[idx] = (bf16)Wr1[k * CE + n];
  }
  for (int idx = t; idx < 43 * 32 * 16; idx += 128) {
    int e = idx & 15, ln = (idx >> 4) & 31, tile = idx >> 9;
    int k = 16 * (ln >> 4) + e, n = tile * 16 + (ln & 15);
    sWr2f[idx] = (bf16)((n < RTOT) ? Wr2[k * RTOT + n] : 0.f);
  }
  for (int idx = t; idx < RPAD; idx += 128) sbr2[idx] = (idx < RTOT) ? br2[idx] : 0.f;
  if (t < CE) {
    sbe1[t] = be1[t]; sbe2[t] = be2[t]; sbr1[t] = br1[t];
    sge1[t] = g_e1[t]; sbel1[t] = b_e1[t];
    sge2[t] = g_e2[t]; sbel2[t] = b_e2[t];
  }
  if (t < DPAIR) { sgp[t] = g_pair[t]; sbp[t] = b_pair[t]; }
  __syncthreads();  // the only block-wide barrier: weights become read-only

  bf16* At = sA[wq]; bf16* Ft = sF[wq]; bf16* Wt = sWt[wq]; bf16* Ht = sH[wq];
  float* Sb = sS[wq]; float* f0s = sf0[wq]; float* f1s = sf1[wq];
  float* dh = sdh[wq]; float* pj = spj[wq]; float* Dij = sDij[wq];
  float* nbrv = snbr[wq]; int* jb = sjb[wq];

  // ---- per-row metadata (lanes 0..15, one (i,k) row each) ----
  if (lane < 16) {
    const int k = wq * 16 + lane;
    const int j = Eidx[i * KNB + k];
    jb[lane] = j;
    float ci[3], cj[3];
#pragma unroll
    for (int d = 0; d < 3; ++d) {
      ci[d] = xyz[(i * 3 + 1) * 3 + d];
      cj[d] = xyz[(j * 3 + 1) * 3 + d];
    }
    float dx = cj[0] - ci[0], dy = cj[1] - ci[1], dz = cj[2] - ci[2];
    float d2 = dx * dx + dy * dy + dz * dz;
    Dij[lane] = sqrtf(fmaxf(d2, 1e-8f));
    float nrm = fmaxf(sqrtf(d2), 1e-8f);
    dh[lane * 3 + 0] = dx / nrm;
    dh[lane * 3 + 1] = dy / nrm;
    dh[lane * 3 + 2] = dz / nrm;
    float di = (float)(j - i);
    nbrv[lane] = (fabsf(di) <= 1.0f) ? di : 0.f;
#pragma unroll
    for (int e = 0; e < 9; ++e) {
      int a = e / 3, d = e % 3;
      f1s[lane * 9 + e] = xyz[(j * 3 + a) * 3 + d] - cj[d];
    }
#pragma unroll
    for (int c = 0; c < 3; ++c)
      pj[lane * 3 + c] = f1s[lane * 9 + c * 3 + 0] * dh[lane * 3 + 0] +
                         f1s[lane * 9 + c * 3 + 1] * dh[lane * 3 + 1] +
                         f1s[lane * 9 + c * 3 + 2] * dh[lane * 3 + 2];
  }
  WAVE_LDS_SYNC();

  // ---- gather f0 = node[j] ----
  for (int r = 0; r < 16; ++r) f0s[r * CE + lane] = node[jb[r] * CE + lane];

  // ---- gather pair rows + LayerNorm(128) -> bf16 A tile (fragment order) ----
  for (int r = 0; r < 16; ++r) {
    const float* prow = pair + ((size_t)i * L + (size_t)jb[r]) * DPAIR;
    if (r + 1 < 16)
      __builtin_prefetch(pair + ((size_t)i * L + (size_t)jb[r + 1]) * DPAIR, 0, 1);
    float x0 = prow[lane * 4 + 0], x1 = prow[lane * 4 + 1];
    float x2 = prow[lane * 4 + 2], x3 = prow[lane * 4 + 3];
    float mu = wsum32(x0 + x1 + x2 + x3) * (1.f / 128.f);
    float y0 = x0 - mu, y1 = x1 - mu, y2 = x2 - mu, y3 = x3 - mu;
    float var = wsum32(y0 * y0 + y1 * y1 + y2 * y2 + y3 * y3) * (1.f / 128.f);
    float rs = rsqrtf(var + 1e-5f);
    const int c = lane * 4;
    At[frag_idx(r, c + 0)] = (bf16)(y0 * rs * sgp[c + 0] + sbp[c + 0]);
    At[frag_idx(r, c + 1)] = (bf16)(y1 * rs * sgp[c + 1] + sbp[c + 1]);
    At[frag_idx(r, c + 2)] = (bf16)(y2 * rs * sgp[c + 2] + sbp[c + 2]);
    At[frag_idx(r, c + 3)] = (bf16)(y3 * rs * sgp[c + 3] + sbp[c + 3]);
  }
  WAVE_LDS_SYNC();

  // ---- stage 1: [16x128] @ We1 -> LN -> Ft cols 0..31 ----
  {
    v8f a0 = vzero8(), a1 = vzero8();
#pragma unroll
    for (int kb = 0; kb < 4; ++kb) {
      v16bf a = load_frag(At, kb, lane);
      v16bf b0 = load_frag(sWe1f, kb * 2 + 0, lane);
      v16bf b1 = load_frag(sWe1f, kb * 2 + 1, lane);
      a0 = wmma_bf16(a, b0, a0);
      a1 = wmma_bf16(a, b1, a1);
    }
    dump_acc_bias(Sb, lane, 0, 32, a0, sbe1);
    dump_acc_bias(Sb, lane, 16, 32, a1, sbe1);
  }
  WAVE_LDS_SYNC();
  ln32_rows_to_frag(Sb, sge1, sbel1, Ft, lane);
  // rbf(36) + neighbor(1) + zero pad -> Ft cols 32..95 (fragment order)
  for (int r = 0; r < 16; ++r) {
    float D = Dij[r], nb = nbrv[r];
#pragma unroll
    for (int base = 32; base < F2PAD; base += 32) {
      int col = base + lane;
      float v;
      if (col < 68) {
        int n = col - 32;
        float mu = 20.f * (float)n * (1.f / 35.f);
        float z = (D - mu) * (36.f / 20.f);
        v = expf(-z * z);
      } else if (col == 68) {
        v = nb;
      } else {
        v = 0.f;
      }
      Ft[frag_idx(r, col)] = (bf16)v;
    }
  }
  WAVE_LDS_SYNC();

  // ---- stage 2: [16x96] @ We2 -> LN -> Wt ----
  {
    v8f a0 = vzero8(), a1 = vzero8();
#pragma unroll
    for (int kb = 0; kb < 3; ++kb) {
      v16bf a = load_frag(Ft, kb, lane);
      v16bf b0 = load_frag(sWe2f, kb * 2 + 0, lane);
      v16bf b1 = load_frag(sWe2f, kb * 2 + 1, lane);
      a0 = wmma_bf16(a, b0, a0);
      a1 = wmma_bf16(a, b1, a1);
    }
    dump_acc_bias(Sb, lane, 0, 32, a0, sbe2);
    dump_acc_bias(Sb, lane, 16, 32, a1, sbe2);
  }
  WAVE_LDS_SYNC();
  ln32_rows_to_frag(Sb, sge2, sbel2, Wt, lane);
  WAVE_LDS_SYNC();

  // ---- stage 3: h = relu(w @ Wr1 + br1) -> Ht (bf16, fragment order) ----
  {
    v8f a0 = vzero8(), a1 = vzero8();
    v16bf a = load_frag(Wt, 0, lane);
    v16bf b0 = load_frag(sWr1f, 0, lane);
    v16bf b1 = load_frag(sWr1f, 1, lane);
    a0 = wmma_bf16(a, b0, a0);
    a1 = wmma_bf16(a, b1, a1);
#pragma unroll
    for (int v = 0; v < 8; ++v) {
      const int row = v + 8 * half;
      Ht[frag_idx(row, r16)] = (bf16)fmaxf(a0[v] + sbr1[r16], 0.f);
      Ht[frag_idx(row, 16 + r16)] = (bf16)fmaxf(a1[v] + sbr1[16 + r16], 0.f);
    }
  }
  WAVE_LDS_SYNC();

  // ---- stage 4: R = h @ Wr2 tile-by-tile, contracted on the fly ----
  float m0a[16], t1a[3], m1a[9];
#pragma unroll
  for (int o = 0; o < 16; ++o) m0a[o] = 0.f;
  t1a[0] = t1a[1] = t1a[2] = 0.f;
#pragma unroll
  for (int e = 0; e < 9; ++e) m1a[e] = 0.f;

  const v16bf aH = load_frag(Ht, 0, lane);

  // region A: R00 cols [0,512) -> m0[o] += R00[o,:] . f0
#pragma unroll
  for (int o = 0; o < 16; ++o) {
    v8f a0 = vzero8(), a1 = vzero8();
    v16bf b0 = load_frag(sWr2f, 2 * o + 0, lane);
    v16bf b1 = load_frag(sWr2f, 2 * o + 1, lane);
    a0 = wmma_bf16(aH, b0, a0);
    a1 = wmma_bf16(aH, b1, a1);
    dump_acc(Sb, lane, 0, 32, a0);
    dump_acc(Sb, lane, 16, 32, a1);
    WAVE_LDS_SYNC();
    float p = 0.f;
#pragma unroll
    for (int ii = 0; ii < 16; ++ii) {
      int c = half * 16 + ii;
      p += (Sb[r16 * 32 + c] + sbr2[o * 32 + c]) * f0s[r16 * CE + c];
    }
    m0a[o] += p;
    WAVE_LDS_SYNC();
  }

  // region B: R10 cols [512,560) -> m0[o] += R10[o,:] . proj
  {
#pragma unroll
    for (int tt = 0; tt < 3; ++tt) {
      v8f a0 = vzero8();
      v16bf b0 = load_frag(sWr2f, 32 + tt, lane);
      a0 = wmma_bf16(aH, b0, a0);
      dump_acc(Sb, lane, tt * 16, 48, a0);
    }
    WAVE_LDS_SYNC();
    if (half == 0) {
      int r = lane;
#pragma unroll
      for (int o = 0; o < 16; ++o) {
        float p = 0.f;
#pragma unroll
        for (int c = 0; c < 3; ++c) {
          int u = o * 3 + c;
          p += (Sb[r * 48 + u] + sbr2[512 + u]) * pj[r * 3 + c];
        }
        m0a[o] += p;
      }
    }
    WAVE_LDS_SYNC();
  }

  // region C: R01 cols [560,656) -> t1[o] += R01[o,:] . f0
  {
#pragma unroll
    for (int tt = 0; tt < 6; ++tt) {
      v8f a0 = vzero8();
      v16bf b0 = load_frag(sWr2f, 35 + tt, lane);
      a0 = wmma_bf16(aH, b0, a0);
      dump_acc(Sb, lane, tt * 16, 96, a0);
    }
    WAVE_LDS_SYNC();
    if (half == 0) {
      int r = lane;
#pragma unroll
      for (int o = 0; o < 3; ++o) {
        float p = 0.f;
#pragma unroll
        for (int ii = 0; ii < 32; ++ii) {
          int u = o * 32 + ii;
          p += (Sb[r * 96 + u] + sbr2[560 + u]) * f0s[r * CE + ii];
        }
        t1a[o] += p;
      }
    }
    WAVE_LDS_SYNC();
  }

  // region D: R11 cols [656,665) and R11b cols [665,674)
  {
#pragma unroll
    for (int tt = 0; tt < 2; ++tt) {
      v8f a0 = vzero8();
      v16bf b0 = load_frag(sWr2f, 41 + tt, lane);
      a0 = wmma_bf16(aH, b0, a0);
      dump_acc(Sb, lane, tt * 16, 32, a0);
    }
    WAVE_LDS_SYNC();
    if (half == 0) {
      int r = lane;
#pragma unroll
      for (int o = 0; o < 3; ++o) {
#pragma unroll
        for (int c = 0; c < 3; ++c) {
          int u = o * 3 + c;
          float v11 = Sb[r * 32 + u] + sbr2[656 + u];
          float v11b = Sb[r * 32 + 9 + u] + sbr2[665 + u];
#pragma unroll
          for (int d = 0; d < 3; ++d) m1a[o * 3 + d] += v11 * f1s[r * 9 + c * 3 + d];
          t1a[o] += v11b * pj[r * 3 + c];
        }
      }
    }
    WAVE_LDS_SYNC();
  }

  // combine lane halves (row r lives in lanes r and r+16)
#pragma unroll
  for (int o = 0; o < 16; ++o) m0a[o] += __shfl_down(m0a[o], 16, 32);
#pragma unroll
  for (int o = 0; o < 3; ++o) t1a[o] += __shfl_down(t1a[o], 16, 32);
#pragma unroll
  for (int e = 0; e < 9; ++e) m1a[e] += __shfl_down(m1a[e], 16, 32);

  // per-wave reduction over its 16 rows -> workspace partials (deterministic)
  if (lane < 16) {
#pragma unroll
    for (int o = 0; o < 16; ++o) Sb[lane * 16 + o] = m0a[o];
  }
  WAVE_LDS_SYNC();
  if (lane < 16) {
    float s = 0.f;
    for (int r = 0; r < 16; ++r) s += Sb[r * 16 + lane];
    pm0[(i * 4 + wq) * 16 + lane] = s;
  }
  WAVE_LDS_SYNC();
  if (lane < 16) {
#pragma unroll
    for (int e = 0; e < 9; ++e)
      Sb[lane * 9 + e] = t1a[e / 3] * dh[lane * 3 + (e % 3)] + m1a[e];
  }
  WAVE_LDS_SYNC();
  if (lane < 9) {
    float s = 0.f;
    for (int r = 0; r < 16; ++r) s += Sb[r * 9 + lane];
    pm1[(i * 4 + wq) * 9 + lane] = s;
  }
}

// ======================= K4: finalize outputs =======================
__global__ __launch_bounds__(32) void finalize_kernel(const float* __restrict__ pm0,
                                                      const float* __restrict__ pm1,
                                                      const float* __restrict__ xyz,
                                                      float* __restrict__ out) {
  const int i = blockIdx.x, lane = threadIdx.x;
  __shared__ float offs[9];
  if (lane < 16) {
    float s = 0.f;
#pragma unroll
    for (int w = 0; w < 4; ++w) s += pm0[(i * 4 + w) * 16 + lane];
    out[9216 + i * 16 + lane] = s * (1.f / 64.f);  // state_out
  }
  if (lane < 9) {
    float s = 0.f;
#pragma unroll
    for (int w = 0; w < 4; ++w) s += pm1[(i * 4 + w) * 9 + lane];
    offs[lane] = s * (1.f / 64.f);  // offset[i][a][d]
  }
  __syncthreads();
  if (lane < 9) {
    const int a = lane / 3, d = lane % 3;
    const float ca = xyz[(i * 3 + 1) * 3 + d];
    const float CAn = ca + offs[3 + d];
    out[i * 9 + a * 3 + d] = (a == 1) ? CAn : (CAn + offs[a * 3 + d]);  // [N, CA, C]
  }
}

// ======================= host glue =======================
extern "C" void kernel_launch(void* const* d_in, const int* in_sizes, int n_in,
                              void* d_out, int out_size, void* d_ws, size_t ws_size,
                              hipStream_t stream) {
  const float* msa = (const float*)d_in[0];
  const float* pair = (const float*)d_in[1];
  const float* xyz = (const float*)d_in[2];
  const float* state = (const float*)d_in[3];
  const float* seq1hot = (const float*)d_in[4];
  // d_in[5] = idx (arange), d_in[6] = top_k (=64): baked as compile-time constants
  const float* g_msa = (const float*)d_in[7];
  const float* b_msa = (const float*)d_in[8];
  const float* g_pair = (const float*)d_in[9];
  const float* b_pair = (const float*)d_in[10];
  const float* g_state = (const float*)d_in[11];
  const float* b_state = (const float*)d_in[12];
  const float* g_node = (const float*)d_in[13];
  const float* b_node = (const float*)d_in[14];
  const float* g_e1 = (const float*)d_in[15];
  const float* b_e1 = (const float*)d_in[16];
  const float* g_e2 = (const float*)d_in[17];
  const float* b_e2 = (const float*)d_in[18];
  const float* Wx = (const float*)d_in[19];
  const float* bx = (const float*)d_in[20];
  const float* We1 = (const float*)d_in[21];
  const float* be1 = (const float*)d_in[22];
  const float* We2 = (const float*)d_in[23];
  const float* be2 = (const float*)d_in[24];
  const float* Wr1 = (const float*)d_in[25];
  const float* br1 = (const float*)d_in[26];
  const float* Wr2 = (const float*)d_in[27];
  const float* br2 = (const float*)d_in[28];

  char* ws = (char*)d_ws;
  float* nodeW = (float*)ws;                              // 1024*32 f32
  int* EidxW = (int*)(ws + 131072);                       // 1024*64 i32
  float* pm0 = (float*)(ws + 131072 + 262144);            // 1024*4*16 f32
  float* pm1 = (float*)(ws + 131072 + 262144 + 262144);   // 1024*4*9 f32
  float* outF = (float*)d_out;

  node_kernel<<<cfg::L, 32, 0, stream>>>(msa, state, seq1hot, g_msa, b_msa, g_state,
                                         b_state, Wx, bx, g_node, b_node, nodeW);
  topk_kernel<<<cfg::L, 256, 0, stream>>>(xyz, EidxW);
  pair_gemm_kernel<<<cfg::L, 128, 0, stream>>>(
      pair, xyz, nodeW, EidxW, g_pair, b_pair, We1, be1, g_e1, b_e1, We2, be2, g_e2,
      b_e2, Wr1, br1, Wr2, br2, pm0, pm1);
  finalize_kernel<<<cfg::L, 32, 0, stream>>>(pm0, pm1, xyz, outF);

  (void)in_sizes; (void)n_in; (void)out_size; (void)ws_size;
}
